// AttentionBlock_32908039422393
// MI455X (gfx1250) — matrix operand order repaired and checked by reading, not compiled, not run on hardware
//
#include <hip/hip_runtime.h>
#include <hip/hip_bf16.h>

// ---------------------------------------------------------------------------
// CDNA5 (gfx1250) fused attention: QKV proj -> scores -> softmax -> P@V
// All GEMMs via v_wmma_f32_16x16x32_bf16 (wave32).
// Wave tile 32x64 (2 A-frags x 4 B-frags, 8 f32 accumulators) ->
// 12 b128 loads per 8 WMMAs per K-step. Block macro-tile 128x128 (8 waves).
// ---------------------------------------------------------------------------

typedef __attribute__((ext_vector_type(16))) __bf16 v16bf;
typedef __attribute__((ext_vector_type(8)))  float  v8f;

union FragU { v16bf v; uint4 q[2]; };

__device__ __forceinline__ unsigned short f2bf(float f) {
    union { float f; unsigned u; } x; x.f = f;
    unsigned r = x.u + 0x7FFFu + ((x.u >> 16) & 1u);   // round-to-nearest-even
    return (unsigned short)(r >> 16);
}

// A-matrix 16x32 bf16 fragment (contiguous k order, same as B): lane L holds row L%16;
// halfs[i] = A[row, k0 + (L/16)*16 + i]  (32 contiguous bytes per lane)
__device__ __forceinline__ v16bf load_a_frag(const unsigned short* base, int ld,
                                             int row0, int k0, int lane) {
    int row = row0 + (lane & 15);
    int kk  = k0 + ((lane >> 4) << 4);
    const uint4* p = (const uint4*)(base + (size_t)row * ld + kk);
    FragU f; f.q[0] = p[0]; f.q[1] = p[1];
    return f.v;
}

// B-matrix 32x16 bf16 fragment: lane L holds column L%16;
// halfs[i] = Brow[col][k0 + (L/16)*16 + i]  (32 contiguous bytes per lane)
__device__ __forceinline__ v16bf load_b_frag(const unsigned short* base, int ld,
                                             int col0, int k0, int lane) {
    int col = col0 + (lane & 15);
    int kk  = k0 + ((lane >> 4) << 4);
    const uint4* p = (const uint4*)(base + (size_t)col * ld + kk);
    FragU f; f.q[0] = p[0]; f.q[1] = p[1];
    return f.v;
}

#define WMMA_BF16(A, B, C) \
    __builtin_amdgcn_wmma_f32_16x16x32_bf16(false, (A), false, (B), (short)0, (C), false, false)

// ------------------------- kernel 1: fp32 -> bf16 --------------------------
__global__ __launch_bounds__(256) void convert_bf16_kernel(
    const float* __restrict__ X, const float* __restrict__ Wq,
    const float* __restrict__ Wk, const float* __restrict__ Wv,
    unsigned short* __restrict__ Xb, unsigned short* __restrict__ Wb)
{
    const size_t nX = (size_t)8192 * 1024;
    const size_t nW = (size_t)1024 * 1024;
    size_t stride = (size_t)gridDim.x * blockDim.x;
    for (size_t idx = (size_t)blockIdx.x * blockDim.x + threadIdx.x;
         idx < nX + 3 * nW; idx += stride) {
        if (idx < nX) {
            Xb[idx] = f2bf(X[idx]);
        } else {
            size_t j = idx - nX;
            const float* W = (j < nW) ? Wq : (j < 2 * nW ? Wk : Wv);
            size_t jj = (j < nW) ? j : (j < 2 * nW ? j - nW : j - 2 * nW);
            Wb[j] = f2bf(W[jj]);
        }
    }
}

// --------------------- shared wave-level GEMM core -------------------------
// Computes a 32x64 tile: acc[i][j] += A(rows rowBase+16i) x B(cols colBase+16j)
// over K = [0, kdim) with leading dims lda/ldb (bf16 elements).
__device__ __forceinline__ void gemm_wave_32x64(
    const unsigned short* __restrict__ Ap, int lda,
    const unsigned short* __restrict__ Bp, int ldb,
    int rowBase, int colBase, int kdim, int lane, v8f acc[2][4])
{
    for (int k = 0; k < kdim; k += 32) {
        v16bf A0 = load_a_frag(Ap, lda, rowBase,      k, lane);
        v16bf A1 = load_a_frag(Ap, lda, rowBase + 16, k, lane);
        v16bf B0 = load_b_frag(Bp, ldb, colBase,      k, lane);
        v16bf B1 = load_b_frag(Bp, ldb, colBase + 16, k, lane);
        v16bf B2 = load_b_frag(Bp, ldb, colBase + 32, k, lane);
        v16bf B3 = load_b_frag(Bp, ldb, colBase + 48, k, lane);
        acc[0][0] = WMMA_BF16(A0, B0, acc[0][0]);
        acc[0][1] = WMMA_BF16(A0, B1, acc[0][1]);
        acc[0][2] = WMMA_BF16(A0, B2, acc[0][2]);
        acc[0][3] = WMMA_BF16(A0, B3, acc[0][3]);
        acc[1][0] = WMMA_BF16(A1, B0, acc[1][0]);
        acc[1][1] = WMMA_BF16(A1, B1, acc[1][1]);
        acc[1][2] = WMMA_BF16(A1, B2, acc[1][2]);
        acc[1][3] = WMMA_BF16(A1, B3, acc[1][3]);
    }
}

// --------------------- kernel 2: QKV projection GEMM -----------------------
// y[m,d] = sum_k X[m,k] * W[d,k] + b[d];  z = 0/1/2 -> Q/K/V
// Q scaled by 1/sqrt(1024)=1/32 (exact in bf16); V stored transposed Vt[b][d][m].
__global__ __launch_bounds__(256) void qkv_gemm_kernel(
    const unsigned short* __restrict__ Xb, const unsigned short* __restrict__ Wb,
    const float* __restrict__ bq, const float* __restrict__ bk, const float* __restrict__ bv,
    unsigned short* __restrict__ Qb, unsigned short* __restrict__ Kb,
    unsigned short* __restrict__ Vt)
{
    const int lane = threadIdx.x & 31;
    const int wave = threadIdx.x >> 5;
    const int rowBase = blockIdx.y * 128 + (wave >> 1) * 32;   // M macro 128
    const int colBase = blockIdx.x * 128 + (wave & 1) * 64;    // D macro 128
    const int z = blockIdx.z;
    const unsigned short* W = Wb + (size_t)z * 1024 * 1024;
    const float* bias = (z == 0) ? bq : (z == 1) ? bk : bv;

    v8f acc[2][4] = {};
    gemm_wave_32x64(Xb, 1024, W, 1024, rowBase, colBase, 1024, lane, acc);

    const float qscale = 0.03125f;   // 1/sqrt(1024)
    for (int i = 0; i < 2; ++i)
    for (int j = 0; j < 4; ++j)
    for (int r = 0; r < 8; ++r) {
        int m = rowBase + i * 16 + ((lane >> 4) << 3) + r;   // C/D: M = r + 8*(lane/16)
        int n = colBase + j * 16 + (lane & 15);
        float val = acc[i][j][r] + bias[n];
        if (z == 0) {
            Qb[(size_t)m * 1024 + n] = f2bf(val * qscale);
        } else if (z == 1) {
            Kb[(size_t)m * 1024 + n] = f2bf(val);
        } else {
            int bb = m >> 11, mm = m & 2047;                 // batch, row-in-batch
            Vt[(((size_t)bb << 10) + n) * 2048 + mm] = f2bf(val);
        }
    }
}

// --------------------- kernel 3: S = Q @ K^T (fp32, L2-resident) -----------
__global__ __launch_bounds__(256) void scores_gemm_kernel(
    const unsigned short* __restrict__ Qb, const unsigned short* __restrict__ Kb,
    float* __restrict__ Sf)
{
    const int lane = threadIdx.x & 31;
    const int wave = threadIdx.x >> 5;
    const int bz = blockIdx.z;
    const unsigned short* Qp = Qb + (size_t)bz * 2048 * 1024;
    const unsigned short* Kp = Kb + (size_t)bz * 2048 * 1024;
    float* Sp = Sf + (size_t)bz * 2048 * 2048;
    const int rowBase = blockIdx.y * 128 + (wave >> 1) * 32;
    const int colBase = blockIdx.x * 128 + (wave & 1) * 64;

    v8f acc[2][4] = {};
    gemm_wave_32x64(Qp, 1024, Kp, 1024, rowBase, colBase, 1024, lane, acc);

    for (int i = 0; i < 2; ++i)
    for (int j = 0; j < 4; ++j)
    for (int r = 0; r < 8; ++r) {
        int m = rowBase + i * 16 + ((lane >> 4) << 3) + r;
        int n = colBase + j * 16 + (lane & 15);
        Sp[(size_t)m * 2048 + n] = acc[i][j][r];
    }
}

// --------------------- kernel 4: row softmax -------------------------------
// One block per row (8192 rows); writes fp32 attention (d_out) + bf16 P (ws).
__global__ __launch_bounds__(256) void softmax_kernel(
    const float* __restrict__ Sf, float* __restrict__ attn,
    unsigned short* __restrict__ Pb)
{
    const int row = blockIdx.x;                 // b*2048 + n
    const int tid = threadIdx.x;
    const float* s = Sf + (size_t)row * 2048;

    float vals[8];
    float m = -3.4e38f;
    for (int i = 0; i < 8; ++i) { vals[i] = s[tid + (i << 8)]; m = fmaxf(m, vals[i]); }
    for (int off = 16; off; off >>= 1) m = fmaxf(m, __shfl_xor(m, off, 32));

    __shared__ float red[8];
    if ((tid & 31) == 0) red[tid >> 5] = m;
    __syncthreads();
    float bm = red[0];
    for (int w = 1; w < 8; ++w) bm = fmaxf(bm, red[w]);

    float lsum = 0.f;
    for (int i = 0; i < 8; ++i) { vals[i] = __expf(vals[i] - bm); lsum += vals[i]; }
    for (int off = 16; off; off >>= 1) lsum += __shfl_xor(lsum, off, 32);
    __syncthreads();
    if ((tid & 31) == 0) red[tid >> 5] = lsum;
    __syncthreads();
    float tot = 0.f;
    for (int w = 0; w < 8; ++w) tot += red[w];
    float inv = 1.0f / tot;

    float* arow = attn + (size_t)row * 2048;
    unsigned short* prow = Pb + (size_t)row * 2048;
    for (int i = 0; i < 8; ++i) {
        float p = vals[i] * inv;
        arow[tid + (i << 8)] = p;
        prow[tid + (i << 8)] = f2bf(p);
    }
}

// --------------------- kernel 5: out = P @ V -------------------------------
__global__ __launch_bounds__(256) void av_gemm_kernel(
    const unsigned short* __restrict__ Pb, const unsigned short* __restrict__ Vt,
    float* __restrict__ outp)
{
    const int lane = threadIdx.x & 31;
    const int wave = threadIdx.x >> 5;
    const int bz = blockIdx.z;
    const unsigned short* Pp = Pb + (size_t)bz * 2048 * 2048;   // [2048 rows][2048 k]
    const unsigned short* Vp = Vt + (size_t)bz * 1024 * 2048;   // [1024 cols][2048 k]
    const int rowBase = blockIdx.y * 128 + (wave >> 1) * 32;
    const int colBase = blockIdx.x * 128 + (wave & 1) * 64;

    v8f acc[2][4] = {};
    gemm_wave_32x64(Pp, 2048, Vp, 2048, rowBase, colBase, 2048, lane, acc);

    for (int i = 0; i < 2; ++i)
    for (int j = 0; j < 4; ++j)
    for (int r = 0; r < 8; ++r) {
        int m = rowBase + i * 16 + ((lane >> 4) << 3) + r;
        int n = colBase + j * 16 + (lane & 15);
        outp[((size_t)(bz * 2048 + m)) * 1024 + n] = acc[i][j][r];
    }
}

// ---------------------------------------------------------------------------
extern "C" void kernel_launch(void* const* d_in, const int* in_sizes, int n_in,
                              void* d_out, int out_size, void* d_ws, size_t ws_size,
                              hipStream_t stream) {
    const float* X  = (const float*)d_in[0];
    const float* Wq = (const float*)d_in[1];
    const float* bq = (const float*)d_in[2];
    const float* Wk = (const float*)d_in[3];
    const float* bk = (const float*)d_in[4];
    const float* Wv = (const float*)d_in[5];
    const float* bv = (const float*)d_in[6];

    float* out  = (float*)d_out;                       // [4,2048,1024]
    float* attn = out + (size_t)4 * 2048 * 1024;       // [4,2048,2048]

    // workspace layout (bf16 unless noted), ~166 MB total
    unsigned short* Xb = (unsigned short*)d_ws;                 // 8192x1024
    unsigned short* Wb = Xb + (size_t)8192 * 1024;              // 3x1024x1024
    unsigned short* Qb = Wb + (size_t)3 * 1024 * 1024;          // 8192x1024 (pre-scaled)
    unsigned short* Kb = Qb + (size_t)8192 * 1024;              // 8192x1024
    unsigned short* Vt = Kb + (size_t)8192 * 1024;              // 4x[1024][2048] transposed
    float*          Sf = (float*)(Vt + (size_t)4 * 1024 * 2048);// 4x2048x2048 fp32 scores
    unsigned short* Pb = (unsigned short*)(Sf + (size_t)4 * 2048 * 2048); // 4x2048x2048

    convert_bf16_kernel<<<2048, 256, 0, stream>>>(X, Wq, Wk, Wv, Xb, Wb);

    dim3 gq(8, 64, 3);    // D/128, M/128, {Q,K,V}
    qkv_gemm_kernel<<<gq, 256, 0, stream>>>(Xb, Wb, bq, bk, bv, Qb, Kb, Vt);

    dim3 gs(16, 16, 4);   // Ncol/128, Nrow/128, B
    scores_gemm_kernel<<<gs, 256, 0, stream>>>(Qb, Kb, Sf);

    softmax_kernel<<<8192, 256, 0, stream>>>(Sf, attn, Pb);

    dim3 ga(8, 16, 4);    // D/128, Nrow/128, B
    av_gemm_kernel<<<ga, 256, 0, stream>>>(Pb, Vt, out);
}
